// Encoder_18657337934707
// MI455X (gfx1250) — compile-verified, matrix operand order checked by reading, not run on hardware
//
#include <hip/hip_runtime.h>
#include <hip/hip_bf16.h>

// ---------------------------------------------------------------------------
// GAT encoder for MI455X (gfx1250, wave32, WMMA).
// Dense layers: pre-packed bf16 operands -> v_wmma_f32_16x16x32_bf16,
// inner loop = 4x global_load_b128 + 1 WMMA. One 16x16 tile per wave.
// Graph softmax-aggregation: atomic segment max/sum (L2-resident).
// ---------------------------------------------------------------------------

typedef __attribute__((ext_vector_type(8)))  __bf16 v8bf;
typedef __attribute__((ext_vector_type(16))) __bf16 v16bf;
typedef __attribute__((ext_vector_type(8)))  float  v8f;

static __device__ __forceinline__ __bf16 f32_to_bf16(float f) {
  union { float f; unsigned u; } v; v.f = f;
  unsigned r = v.u + 0x7FFFu + ((v.u >> 16) & 1u);   // round-to-nearest-even
  unsigned short h = (unsigned short)(r >> 16);
  return __builtin_bit_cast(__bf16, h);
}

static __device__ __forceinline__ void atomicMaxF(float* addr, float val) {
  // ordered-bits trick; valid with -inf init and mixed-sign updates
  if (val >= 0.0f) atomicMax((int*)addr, __float_as_int(val));
  else             atomicMin((unsigned int*)addr, __float_as_uint(val));
}

// ---------------- operand packing for WMMA ----------------------------------
// A: f32 [N,K] -> bf16 [N,K] (same layout)
__global__ void k_cvt_bf16(const float* __restrict__ in, __bf16* __restrict__ out, int n) {
  int t = blockIdx.x * blockDim.x + threadIdx.x;
  if (t >= n) return;
  out[t] = f32_to_bf16(in[t]);
}

// W: f32 [K,M] -> bf16 [M,K] (transposed so B-fragment loads are contiguous)
__global__ void k_cvt_w_t(const float* __restrict__ W, __bf16* __restrict__ Wt,
                          int K, int M) {
  int t = blockIdx.x * blockDim.x + threadIdx.x;
  if (t >= K * M) return;
  int k = t / M, m = t - k * M;
  Wt[(size_t)m * K + k] = f32_to_bf16(W[t]);
}

// ---------------- dense GEMM: C[N,M] = A[N,K] @ W[K,M] ----------------------
// Requires N % 16 == 0 (true for all launches here: N = 20000).
// CDNA5 16-bit fragment layout: lane group hi = lane>>4 holds
// K = hi*8 + {0..7} and hi*8 + {16..23}  ->  two contiguous b128 loads.
template <int K, int M>
__global__ __launch_bounds__(256)
void k_gemm_wmma(const __bf16* __restrict__ Abf,   // [N,K] bf16 row-major
                 const __bf16* __restrict__ Wt,    // [M,K] bf16 (W^T)
                 float* __restrict__ C, int N) {
  const int lane    = threadIdx.x & 31;
  const int wave    = (int)((blockIdx.x * blockDim.x + threadIdx.x) >> 5);
  const int tilesM  = M >> 4;
  const int tileRow = wave / tilesM;
  const int tileCol = wave - tileRow * tilesM;
  if (tileRow * 16 >= N) return;               // wave-uniform; EXEC all-1 at WMMA

  const int hi = lane >> 4;
  const int lm = lane & 15;
  const __bf16* __restrict__ Ar = Abf + (size_t)(tileRow * 16 + lm) * K + hi * 8;
  const __bf16* __restrict__ Br = Wt  + (size_t)(tileCol * 16 + lm) * K + hi * 8;

  v8f acc = {};
#pragma unroll
  for (int k0 = 0; k0 < K; k0 += 32) {
    v8bf a0 = *(const v8bf*)(Ar + k0);
    v8bf a1 = *(const v8bf*)(Ar + k0 + 16);
    v8bf b0 = *(const v8bf*)(Br + k0);
    v8bf b1 = *(const v8bf*)(Br + k0 + 16);
    v16bf afrag = __builtin_shufflevector(a0, a1, 0,1,2,3,4,5,6,7,8,9,10,11,12,13,14,15);
    v16bf bfrag = __builtin_shufflevector(b0, b1, 0,1,2,3,4,5,6,7,8,9,10,11,12,13,14,15);
    acc = __builtin_amdgcn_wmma_f32_16x16x32_bf16(false, afrag, false, bfrag,
                                                  (short)0, acc, false, false);
  }
  // C/D layout: VGPR r -> row r (lanes 0-15) / r+8 (lanes 16-31), col = lane&15
  float* __restrict__ Cr = C + (size_t)(tileRow * 16 + 8 * hi) * M + tileCol * 16 + lm;
#pragma unroll
  for (int r = 0; r < 8; ++r)
    Cr[(size_t)r * M] = acc[r];
}

// ---------------- Cin==1 "GEMM": outer product (+ optional bias) ------------
__global__ void k_outer1(const float* __restrict__ x, const float* __restrict__ W,
                         const float* __restrict__ b, float* __restrict__ out,
                         int N, int Cc, int hasB) {
  int t = blockIdx.x * blockDim.x + threadIdx.x;
  if (t >= N * Cc) return;
  int n = t / Cc, c = t - n * Cc;
  float v = x[n] * W[c];
  if (hasB) v += b[c];
  out[t] = v;
}

// ---------------- per-node attention dots: wave per node --------------------
__global__ __launch_bounds__(256)
void k_dots(const float* __restrict__ H, const float* __restrict__ as,
            const float* __restrict__ ad, float* __restrict__ ssrc,
            float* __restrict__ sdst, int N, int Cc) {
  int wave = (int)((blockIdx.x * blockDim.x + threadIdx.x) >> 5);
  int lane = threadIdx.x & 31;
  if (wave >= N) return;
  const float* h = H + (size_t)wave * Cc;
  float s0 = 0.f, s1 = 0.f;
  for (int c = lane; c < Cc; c += 32) {
    float hv = h[c];
    s0 += hv * as[c];
    s1 += hv * ad[c];
  }
#pragma unroll
  for (int off = 16; off > 0; off >>= 1) {
    s0 += __shfl_down(s0, off, 32);
    s1 += __shfl_down(s1, off, 32);
  }
  if (lane == 0) { ssrc[wave] = s0; sdst[wave] = s1; }
}

// ---------------- segment softmax over edges --------------------------------
__global__ void k_init_md(float* __restrict__ m, float* __restrict__ d, int N) {
  int t = blockIdx.x * blockDim.x + threadIdx.x;
  if (t >= N) return;
  m[t] = -__builtin_inff();
  d[t] = 0.f;
}

__global__ void k_edge_logits(const int* __restrict__ esrc, const int* __restrict__ edst,
                              const float* __restrict__ ssrc, const float* __restrict__ sdst,
                              float* __restrict__ ev, float* __restrict__ mbuf,
                              int E, int N) {
  int e = blockIdx.x * blockDim.x + threadIdx.x;
  int ET = E + N;
  if (e >= ET) return;
  int s = (e < E) ? esrc[e] : (e - E);       // implicit self loops
  int d = (e < E) ? edst[e] : (e - E);
  float v = ssrc[s] + sdst[d];
  v = (v > 0.f) ? v : 0.2f * v;              // leaky_relu(0.2)
  ev[e] = v;
  atomicMaxF(&mbuf[d], v);
}

__global__ void k_edge_exp(const int* __restrict__ edst, const float* __restrict__ ev,
                           const float* __restrict__ mbuf, float* __restrict__ alpha,
                           float* __restrict__ dbuf, int E, int N) {
  int e = blockIdx.x * blockDim.x + threadIdx.x;
  int ET = E + N;
  if (e >= ET) return;
  int d = (e < E) ? edst[e] : (e - E);
  float a = __expf(ev[e] - mbuf[d]);
  alpha[e] = a;
  atomicAdd(&dbuf[d], a);
}

__global__ void k_edge_norm(const int* __restrict__ edst, float* __restrict__ alpha,
                            const float* __restrict__ dbuf, int E, int N) {
  int e = blockIdx.x * blockDim.x + threadIdx.x;
  int ET = E + N;
  if (e >= ET) return;
  int d = (e < E) ? edst[e] : (e - E);
  alpha[e] = alpha[e] / dbuf[d];
}

// ---------------- weighted scatter-add: out[dst] += alpha * h[src] ----------
__global__ void k_edge_scatter(const int* __restrict__ esrc, const int* __restrict__ edst,
                               const float* __restrict__ alpha, const float* __restrict__ H,
                               float* __restrict__ out, int E, int N, int Cc) {
  int t = blockIdx.x * blockDim.x + threadIdx.x;
  int chunks = Cc >> 2;
  int ET = E + N;
  if (t >= ET * chunks) return;
  int e  = t / chunks;
  int c4 = (t - e * chunks) << 2;
  int s = (e < E) ? esrc[e] : (e - E);
  int d = (e < E) ? edst[e] : (e - E);
  float w = alpha[e];
  const float4 hv = *(const float4*)(H + (size_t)s * Cc + c4);
  float* o = out + (size_t)d * Cc + c4;
  atomicAdd(o + 0, w * hv.x);
  atomicAdd(o + 1, w * hv.y);
  atomicAdd(o + 2, w * hv.z);
  atomicAdd(o + 3, w * hv.w);
}

// ---------------- elementwise tails -----------------------------------------
__global__ void k_zero(float* __restrict__ p, int n) {
  int t = blockIdx.x * blockDim.x + threadIdx.x;
  if (t < n) p[t] = 0.f;
}

__global__ void k_bias_act(float* __restrict__ out, const float* __restrict__ b,
                           int N, int Cc, int relu) {
  int t = blockIdx.x * blockDim.x + threadIdx.x;
  if (t >= N * Cc) return;
  float v = out[t] + b[t % Cc];
  if (relu) v = v > 0.f ? v : 0.f;
  out[t] = v;
}

__global__ void k_combine_leaky(const float* __restrict__ x1, const float* __restrict__ g,
                                const float* __restrict__ b, float* __restrict__ x2,
                                int n, int Cc) {
  int t = blockIdx.x * blockDim.x + threadIdx.x;
  if (t >= n) return;
  float v = x1[t] + g[t] + b[t % Cc];
  x2[t] = (v > 0.f) ? v : 0.01f * v;
}

__global__ void k_final_sigmoid(const float* __restrict__ x2, const float* __restrict__ g,
                                const float* __restrict__ b, float* __restrict__ out,
                                int n, int Cc) {
  int t = blockIdx.x * blockDim.x + threadIdx.x;
  if (t >= n) return;
  float v = x2[t] + g[t] + b[t % Cc];
  out[t] = 1.f / (1.f + __expf(-v));
}

// ---------------------------------------------------------------------------
// Host-side orchestration
// ---------------------------------------------------------------------------
static inline int cdiv(int a, int b) { return (a + b - 1) / b; }

static void launch_gemm(const float* A, const float* W, float* C,
                        int N, int K, int M, __bf16* Abf, __bf16* Wtbf,
                        hipStream_t stream) {
  k_cvt_bf16<<<cdiv(N * K, 256), 256, 0, stream>>>(A, Abf, N * K);
  k_cvt_w_t<<<cdiv(K * M, 256), 256, 0, stream>>>(W, Wtbf, K, M);
  int waves  = (N / 16) * (M / 16);
  int blocks = cdiv(waves, 8);                 // 8 waves / 256-thread block
  if (K == 128 && M == 128)
    k_gemm_wmma<128, 128><<<blocks, 256, 0, stream>>>(Abf, Wtbf, C, N);
  else if (K == 128 && M == 256)
    k_gemm_wmma<128, 256><<<blocks, 256, 0, stream>>>(Abf, Wtbf, C, N);
  else
    k_gemm_wmma<256, 256><<<blocks, 256, 0, stream>>>(Abf, Wtbf, C, N);
}

static void gat_conv(const float* in, int Cin, int Cout,
                     const float* W, const float* as, const float* ad, const float* bias,
                     const int* esrc, const int* edst, int E, int N,
                     float* h, float* out,
                     float* ssrc, float* sdst, float* mbuf, float* dbuf,
                     float* ev, float* alpha, __bf16* Abf, __bf16* Wtbf,
                     int relu, hipStream_t stream) {
  // 1. h = in @ W
  if (Cin == 1) {
    k_outer1<<<cdiv(N * Cout, 256), 256, 0, stream>>>(in, W, nullptr, h, N, Cout, 0);
  } else {
    launch_gemm(in, W, h, N, Cin, Cout, Abf, Wtbf, stream);
  }
  // 2. per-node attention scalars
  k_dots<<<cdiv(N * 32, 256), 256, 0, stream>>>(h, as, ad, ssrc, sdst, N, Cout);
  // 3. segment softmax
  int ET = E + N;
  k_init_md<<<cdiv(N, 256), 256, 0, stream>>>(mbuf, dbuf, N);
  k_edge_logits<<<cdiv(ET, 256), 256, 0, stream>>>(esrc, edst, ssrc, sdst, ev, mbuf, E, N);
  k_edge_exp<<<cdiv(ET, 256), 256, 0, stream>>>(edst, ev, mbuf, alpha, dbuf, E, N);
  k_edge_norm<<<cdiv(ET, 256), 256, 0, stream>>>(edst, alpha, dbuf, E, N);
  // 4. aggregate
  k_zero<<<cdiv(N * Cout, 256), 256, 0, stream>>>(out, N * Cout);
  int scat = ET * (Cout / 4);
  k_edge_scatter<<<cdiv(scat, 256), 256, 0, stream>>>(esrc, edst, alpha, h, out, E, N, Cout);
  // 5. bias (+ReLU between layers)
  k_bias_act<<<cdiv(N * Cout, 256), 256, 0, stream>>>(out, bias, N, Cout, relu);
}

extern "C" void kernel_launch(void* const* d_in, const int* in_sizes, int n_in,
                              void* d_out, int out_size, void* d_ws, size_t ws_size,
                              hipStream_t stream) {
  (void)n_in; (void)out_size; (void)ws_size;
  const int N = in_sizes[0];          // 20000  (multiple of 16)
  const int E = in_sizes[1] / 2;      // 320000
  const int C = 256;

  const float* x    = (const float*)d_in[0];
  const int*   ei   = (const int*)d_in[1];
  const int*   esrc = ei;
  const int*   edst = ei + E;

  // flattened param order: g1[3]{W,as,ad,b}, g2[3]{W,as,ad,b}, lin1{W,b}, lin2{W,b}, lin3{W,b}
  const float* g1W[3]  = {(const float*)d_in[2],  (const float*)d_in[6],  (const float*)d_in[10]};
  const float* g1as[3] = {(const float*)d_in[3],  (const float*)d_in[7],  (const float*)d_in[11]};
  const float* g1ad[3] = {(const float*)d_in[4],  (const float*)d_in[8],  (const float*)d_in[12]};
  const float* g1b[3]  = {(const float*)d_in[5],  (const float*)d_in[9],  (const float*)d_in[13]};
  const float* g2W[3]  = {(const float*)d_in[14], (const float*)d_in[18], (const float*)d_in[22]};
  const float* g2as[3] = {(const float*)d_in[15], (const float*)d_in[19], (const float*)d_in[23]};
  const float* g2ad[3] = {(const float*)d_in[16], (const float*)d_in[20], (const float*)d_in[24]};
  const float* g2b[3]  = {(const float*)d_in[17], (const float*)d_in[21], (const float*)d_in[25]};
  const float* lin1W = (const float*)d_in[26]; const float* lin1b = (const float*)d_in[27];
  const float* lin2W = (const float*)d_in[28]; const float* lin2b = (const float*)d_in[29];
  const float* lin3W = (const float*)d_in[30]; const float* lin3b = (const float*)d_in[31];

  // workspace carve-up (~116 MB)
  float* ws    = (float*)d_ws;
  float* bufA  = ws;                         // N*256  (GEMM output h)
  float* bufB  = bufA + (size_t)N * C;       // N*256
  float* bufC  = bufB + (size_t)N * C;       // N*256
  float* x1b   = bufC + (size_t)N * C;       // N*256
  float* x2b   = x1b  + (size_t)N * C;       // N*256
  float* ssrc  = x2b  + (size_t)N * C;       // N
  float* sdst  = ssrc + N;                   // N
  float* mbuf  = sdst + N;                   // N
  float* dbuf  = mbuf + N;                   // N
  float* ev    = dbuf + N;                   // E+N
  float* alpha = ev   + (E + N);             // E+N
  __bf16* Abf  = (__bf16*)(alpha + (E + N)); // N*256 bf16
  __bf16* Wtbf = Abf + (size_t)N * C;        // 256*256 bf16

  // x1 = x @ lin1.W + lin1.b
  k_outer1<<<cdiv(N * C, 256), 256, 0, stream>>>(x, lin1W, lin1b, x1b, N, C, 1);

  // ---- gat1 stack: 1 -> 128 -> 128 -> 256 (ReLU between layers) ----
  gat_conv(x,    1,   128, g1W[0], g1as[0], g1ad[0], g1b[0], esrc, edst, E, N,
           bufA, bufB, ssrc, sdst, mbuf, dbuf, ev, alpha, Abf, Wtbf, 1, stream);
  gat_conv(bufB, 128, 128, g1W[1], g1as[1], g1ad[1], g1b[1], esrc, edst, E, N,
           bufA, bufC, ssrc, sdst, mbuf, dbuf, ev, alpha, Abf, Wtbf, 1, stream);
  gat_conv(bufC, 128, 256, g1W[2], g1as[2], g1ad[2], g1b[2], esrc, edst, E, N,
           bufA, bufB, ssrc, sdst, mbuf, dbuf, ev, alpha, Abf, Wtbf, 0, stream);

  // x2 = leaky_relu(x1 + bufB @ lin2.W + lin2.b, 0.01)
  launch_gemm(bufB, lin2W, bufC, N, C, C, Abf, Wtbf, stream);
  k_combine_leaky<<<cdiv(N * C, 256), 256, 0, stream>>>(x1b, bufC, lin2b, x2b, N * C, C);

  // ---- gat2 stack: 256 -> 256 -> 256 -> 256 ----
  gat_conv(x2b,  256, 256, g2W[0], g2as[0], g2ad[0], g2b[0], esrc, edst, E, N,
           bufA, bufB, ssrc, sdst, mbuf, dbuf, ev, alpha, Abf, Wtbf, 1, stream);
  gat_conv(bufB, 256, 256, g2W[1], g2as[1], g2ad[1], g2b[1], esrc, edst, E, N,
           bufA, bufC, ssrc, sdst, mbuf, dbuf, ev, alpha, Abf, Wtbf, 1, stream);
  gat_conv(bufC, 256, 256, g2W[2], g2as[2], g2ad[2], g2b[2], esrc, edst, E, N,
           bufA, bufB, ssrc, sdst, mbuf, dbuf, ev, alpha, Abf, Wtbf, 0, stream);

  // out = sigmoid(x2 + bufB @ lin3.W + lin3.b)
  launch_gemm(bufB, lin3W, bufC, N, C, C, Abf, Wtbf, stream);
  k_final_sigmoid<<<cdiv(N * C, 256), 256, 0, stream>>>(x2b, bufC, lin3b,
                                                        (float*)d_out, N * C, C);
}
